// MatchingNet_83958020702528
// MI455X (gfx1250) — matrix operand order repaired and checked by reading, not compiled, not run on hardware
//
#include <hip/hip_runtime.h>
#include <math.h>

typedef _Float16 f16;
typedef __attribute__((ext_vector_type(16))) _Float16 v16h;
typedef __attribute__((ext_vector_type(8)))  _Float16 v8h;
typedef __attribute__((ext_vector_type(8)))  float    v8f;

__device__ __forceinline__ float sigm(float x) { return 1.0f / (1.0f + expf(-x)); }

__device__ __forceinline__ v8f wmma_f16(v16h a, v16h b, v8f c) {
    return __builtin_amdgcn_wmma_f32_16x16x32_f16(false, a, false, b, (short)0, c, false, false);
}

// ---------------------------------------------------------------------------
// Prep kernels
// ---------------------------------------------------------------------------

// Fold conv bias + inference BN into per-channel scale/bias for all 4 blocks.
// scb layout: scale at [blk*64+o], bias at [256 + blk*64 + o]
__global__ __launch_bounds__(256) void k_prep_sb(
    const float* b1, const float* g1, const float* be1, const float* m1, const float* v1,
    const float* b2, const float* g2, const float* be2, const float* m2, const float* v2,
    const float* b3, const float* g3, const float* be3, const float* m3, const float* v3,
    const float* b4, const float* g4, const float* be4, const float* m4, const float* v4,
    float* scb)
{
    int t = threadIdx.x;                 // 256 threads
    int blk = t >> 6, o = t & 63;
    const float* bb[4] = {b1, b2, b3, b4};
    const float* gg[4] = {g1, g2, g3, g4};
    const float* be[4] = {be1, be2, be3, be4};
    const float* mm[4] = {m1, m2, m3, m4};
    const float* vv[4] = {v1, v2, v3, v4};
    float inv = gg[blk][o] / sqrtf(vv[blk][o] + 1e-5f);
    scb[blk * 64 + o] = inv;
    scb[256 + blk * 64 + o] = (bb[blk][o] - mm[blk][o]) * inv + be[blk][o];
}

// OIHW f32 [64][64][3][3] -> per-tap Bt f16 [9][O=64][I=64]  (K contiguous)
__global__ __launch_bounds__(256) void k_prep_convw(const float* __restrict__ w, f16* __restrict__ out)
{
    int t = blockIdx.x * 256 + threadIdx.x;   // 36864
    int i = t & 63;
    int o = (t >> 6) & 63;
    int tap = t >> 12;                        // 0..8
    int ky = tap / 3, kx = tap % 3;
    out[t] = (f16)w[o * 27 + i * 9 + ky * 3 + kx];
}

// Transpose [4096][1024] f32 -> [1024][4096] f16 (for coalesced GEMV)
__global__ __launch_bounds__(256) void k_transpose(const float* __restrict__ in, f16* __restrict__ out)
{
    int t = blockIdx.x * 256 + threadIdx.x;   // 4,194,304
    int j = t & 4095;
    int k = t >> 12;
    out[(size_t)k * 4096 + j] = (f16)in[(size_t)j * 1024 + k];
}

// Bc[n][0:2048] = wc_ih[n], Bc[n][2048:3072] = wc_hh[n]  (f16, K contiguous)
__global__ __launch_bounds__(256) void k_prep_fcew(const float* __restrict__ wcih,
                                                   const float* __restrict__ wchh,
                                                   f16* __restrict__ out)
{
    int t = blockIdx.x * 256 + threadIdx.x;   // 12,582,912
    int col = t % 3072;
    int n = t / 3072;
    float v = (col < 2048) ? wcih[(size_t)n * 2048 + col]
                           : wchh[(size_t)n * 1024 + (col - 2048)];
    out[t] = (f16)v;
}

__global__ __launch_bounds__(256) void k_prep_bias(
    const float* bf_ih, const float* bf_hh, const float* br_ih, const float* br_hh,
    const float* bc_ih, const float* bc_hh, float* bF, float* bR, float* bC)
{
    int t = blockIdx.x * 256 + threadIdx.x;   // 12288
    int sel = t >> 12, j = t & 4095;
    const float* pa[3] = {bf_ih, br_ih, bc_ih};
    const float* pb[3] = {bf_hh, br_hh, bc_hh};
    float* po[3] = {bF, bR, bC};
    po[sel][j] = pa[sel][j] + pb[sel][j];
}

// ---------------------------------------------------------------------------
// Encoder
// ---------------------------------------------------------------------------

// Block 1: direct f32 conv (Cin=3) + scale/bias + ReLU + 2x2 maxpool fused.
// Output NHWC f16 [400][32][32][64].
__global__ __launch_bounds__(256) void k_conv1(const float* __restrict__ q,
                                               const float* __restrict__ sup,
                                               const float* __restrict__ w,
                                               const float* __restrict__ scb,
                                               f16* __restrict__ out)
{
    int t = blockIdx.x * 256 + threadIdx.x;   // 26,214,400
    int oc = t & 63;
    int x2 = (t >> 6) & 31;
    int y2 = (t >> 11) & 31;
    int img = t >> 16;
    const float* in = (img < 300) ? (q + (size_t)img * 3 * 4096)
                                  : (sup + (size_t)(img - 300) * 3 * 4096);
    float sc = scb[oc], bi = scb[256 + oc];
    const float* wr = w + oc * 27;
    float best = -1e30f;
    for (int dy = 0; dy < 2; ++dy)
        for (int dx = 0; dx < 2; ++dx) {
            int Y = 2 * y2 + dy, X = 2 * x2 + dx;
            float acc = 0.f;
            for (int ic = 0; ic < 3; ++ic)
                for (int ky = 0; ky < 3; ++ky) {
                    int iy = Y + ky - 1;
                    if (iy < 0 || iy > 63) continue;
                    for (int kx = 0; kx < 3; ++kx) {
                        int ix = X + kx - 1;
                        if (ix < 0 || ix > 63) continue;
                        acc += in[ic * 4096 + iy * 64 + ix] * wr[(ic * 3 + ky) * 3 + kx];
                    }
                }
            best = fmaxf(best, fmaxf(acc * sc + bi, 0.f));
        }
    out[t] = (f16)best;
}

// Blocks 2-4: implicit-GEMM WMMA conv (Cin=64, Cout=64), NHWC f16 in/out.
// One wave per (img, 16-pixel M tile, 16-channel N tile); 18 (tap,khalf)
// steps of v_wmma_f32_16x16x32_f16, software-pipelined by one stage so the
// next operands load while the current WMMA executes. SD is compile-time.
template <int SD, int MT>
__global__ __launch_bounds__(256) void k_conv_wmma(const f16* __restrict__ in,
                                                   const f16* __restrict__ w9,
                                                   const float* __restrict__ scale,
                                                   const float* __restrict__ bias,
                                                   f16* __restrict__ out)
{
    int wid = (blockIdx.x * 256 + threadIdx.x) >> 5;
    int lane = threadIdx.x & 31;
    int nt = wid & 3;
    int mt = (wid >> 2) % MT;       // MT is a power of two -> mask
    int img = wid / (MT << 2);      // power of two -> shift
    int row = lane & 15;            // A row (M), B column (N), D column (N)
    int hsel = lane >> 4;           // lane-half selector

    int p = mt * 16 + row;
    int y = p / SD, x = p % SD;     // compile-time SD -> shift/mask

    v8f acc;
#pragma unroll
    for (int i = 0; i < 8; ++i) acc[i] = 0.f;

    // step t in [0,18): tap = t>>1 (ty=tap/3, tx=tap%3), khalf = t&1
    auto loadA = [&](int t) -> v16h {
        int tap = t >> 1, kh = t & 1;
        int ty = tap / 3, tx = tap % 3;
        int iy = y + ty - 1, ix = x + tx - 1;
        bool ok = ((unsigned)iy < (unsigned)SD) && ((unsigned)ix < (unsigned)SD);
        v8h lo, hi;
        if (ok) {
            const f16* pa = in + ((((size_t)img * SD + iy) * SD + ix) << 6)
                               + kh * 32 + hsel * 16;
            lo = *(const v8h*)pa;
            hi = *(const v8h*)(pa + 8);
        } else {
#pragma unroll
            for (int i = 0; i < 8; ++i) { lo[i] = (f16)0.f; hi[i] = (f16)0.f; }
        }
        return __builtin_shufflevector(lo, hi,
            0, 1, 2, 3, 4, 5, 6, 7, 8, 9, 10, 11, 12, 13, 14, 15);
    };
    auto loadB = [&](int t) -> v16h {
        int tap = t >> 1, kh = t & 1;
        const f16* wb = w9 + ((tap * 64 + (nt * 16 + row)) << 6);
        return *(const v16h*)(wb + kh * 32 + hsel * 16);
    };

    v16h a0 = loadA(0);
    v16h b0 = loadB(0);
#pragma unroll
    for (int t = 0; t < 17; ++t) {
        v16h a1 = loadA(t + 1);
        v16h b1 = loadB(t + 1);
        acc = wmma_f16(a0, b0, acc);
        a0 = a1; b0 = b1;
    }
    acc = wmma_f16(a0, b0, acc);

    int oc = nt * 16 + row;
    float sc = scale[oc], bi = bias[oc];
#pragma unroll
    for (int v = 0; v < 8; ++v) {
        int pp = mt * 16 + v + hsel * 8;
        float r = fmaxf(acc[v] * sc + bi, 0.f);
        out[((((size_t)img * SD + pp / SD) * SD + (pp % SD)) << 6) + oc] = (f16)r;
    }
}

// 2x2 maxpool NHWC f16 (SD compile-time -> shift/mask indexing)
template <int SD>
__global__ __launch_bounds__(256) void k_pool(const f16* __restrict__ in, f16* __restrict__ out)
{
    constexpr int SD2 = SD / 2;
    int t = blockIdx.x * 256 + threadIdx.x;
    int oc = t & 63;
    int r = t >> 6;
    int x = r % SD2; r /= SD2;
    int y = r % SD2;
    int img = r / SD2;
    const f16* base = in + ((size_t)img * SD * SD << 6);
    float m = -1e30f;
#pragma unroll
    for (int dy = 0; dy < 2; ++dy)
#pragma unroll
        for (int dx = 0; dx < 2; ++dx)
            m = fmaxf(m, (float)base[((((2 * y + dy) * SD) + 2 * x + dx) << 6) + oc]);
    out[t] = (f16)m;
}

// NHWC [400][4][4][64] f16 -> emb f32 [400][1024] with reference NCHW flatten order
__global__ __launch_bounds__(256) void k_flatten(const f16* __restrict__ act4, float* __restrict__ emb)
{
    int t = blockIdx.x * 256 + threadIdx.x;   // 409600
    int hidx = t & 1023;
    int img = t >> 10;
    int c = hidx >> 4;
    int y = (hidx >> 2) & 3;
    int x = hidx & 3;
    emb[t] = (float)act4[(((size_t)(img * 4 + y) * 4 + x) << 6) + c];
}

// ---------------------------------------------------------------------------
// Support BiLSTM scan (per-step kernel, 4 batches co-processed, f16 weights)
// ---------------------------------------------------------------------------

__global__ __launch_bounds__(256) void k_zero2(float* a, float* b)
{
    int t = blockIdx.x * 256 + threadIdx.x;   // 8192
    a[t] = 0.f;
    b[t] = 0.f;
}

__global__ __launch_bounds__(256) void k_scan_step(
    const float* __restrict__ emb,
    const f16* __restrict__ WtFih, const f16* __restrict__ WtFhh,
    const f16* __restrict__ WtRih, const f16* __restrict__ WtRhh,
    const float* __restrict__ bF, const float* __restrict__ bR,
    const float* __restrict__ hread, float* __restrict__ hwrite,
    float* __restrict__ cscan, float* __restrict__ hseq, int s)
{
    __shared__ float xs[4][1024];
    __shared__ float hs[4][1024];
    int tid = threadIdx.x;                    // 256
    int dir = blockIdx.x >> 2;                // 8 blocks: (dir, etile)
    int et = blockIdx.x & 3;
    int seq = dir ? (24 - s) : s;

    for (int b = 0; b < 4; ++b) {
        const float* xr = emb + (size_t)(300 + b * 25 + seq) * 1024;
        const float* hr = hread + (size_t)(b * 2 + dir) * 1024;
        for (int k = tid; k < 1024; k += 256) { xs[b][k] = xr[k]; hs[b][k] = hr[k]; }
    }
    __syncthreads();

    int e = et * 256 + tid;
    const f16* Wih = dir ? WtRih : WtFih;
    const f16* Whh = dir ? WtRhh : WtFhh;
    const float* bD = dir ? bR : bF;

    float aI[4], aF[4], aG[4], aO[4];
#pragma unroll
    for (int b = 0; b < 4; ++b) {
        aI[b] = bD[e]; aF[b] = bD[1024 + e]; aG[b] = bD[2048 + e]; aO[b] = bD[3072 + e];
    }
    for (int k = 0; k < 1024; ++k) {
        const f16* wi = Wih + (size_t)k * 4096 + e;
        const f16* wh = Whh + (size_t)k * 4096 + e;
        float wi0 = (float)wi[0], wi1 = (float)wi[1024], wi2 = (float)wi[2048], wi3 = (float)wi[3072];
        float wh0 = (float)wh[0], wh1 = (float)wh[1024], wh2 = (float)wh[2048], wh3 = (float)wh[3072];
#pragma unroll
        for (int b = 0; b < 4; ++b) {
            float xv = xs[b][k], hv = hs[b][k];
            aI[b] += xv * wi0 + hv * wh0;
            aF[b] += xv * wi1 + hv * wh1;
            aG[b] += xv * wi2 + hv * wh2;
            aO[b] += xv * wi3 + hv * wh3;
        }
    }
#pragma unroll
    for (int b = 0; b < 4; ++b) {
        size_t hc = (size_t)(b * 2 + dir) * 1024 + e;
        float c = cscan[hc];
        float cn = sigm(aF[b]) * c + sigm(aI[b]) * tanhf(aG[b]);
        float hn = sigm(aO[b]) * tanhf(cn);
        cscan[hc] = cn;
        hwrite[hc] = hn;
        hseq[((size_t)(b * 2 + dir) * 25 + seq) * 1024 + e] = hn;
    }
}

// ctx = s_emb + hf + hr ; s_norm = ctx / max(||ctx||, 1e-12)
__global__ __launch_bounds__(256) void k_ctxnorm(const float* __restrict__ emb,
                                                 const float* __restrict__ hseq,
                                                 float* __restrict__ ctx,
                                                 float* __restrict__ snorm)
{
    __shared__ float v[1024];
    __shared__ float red[256];
    int bt = blockIdx.x;                      // 100 = (b,t)
    int b = bt / 25, t = bt % 25;
    int tid = threadIdx.x;
    const float* se = emb + (size_t)(300 + b * 25 + t) * 1024;
    const float* hf = hseq + ((size_t)(b * 2) * 25 + t) * 1024;
    const float* hr = hseq + ((size_t)(b * 2 + 1) * 25 + t) * 1024;
    float ss = 0.f;
    for (int k = tid; k < 1024; k += 256) {
        float x = se[k] + hf[k] + hr[k];
        v[k] = x;
        ss += x * x;
    }
    red[tid] = ss;
    __syncthreads();
    for (int o = 128; o > 0; o >>= 1) {
        if (tid < o) red[tid] += red[tid + o];
        __syncthreads();
    }
    float inv = 1.f / fmaxf(sqrtf(red[0]), 1e-12f);
    for (int k = tid; k < 1024; k += 256) {
        ctx[(size_t)bt * 1024 + k] = v[k];
        snorm[(size_t)bt * 1024 + k] = v[k] * inv;
    }
}

// ---------------------------------------------------------------------------
// FCE phase
// ---------------------------------------------------------------------------

// Afce f16 [304][3072]: [q | readout | h]; hcur = q_emb; ccur = 0; pad rows = 0
__global__ __launch_bounds__(256) void k_fce_init(const float* __restrict__ emb,
                                                  f16* __restrict__ A,
                                                  float* __restrict__ hcur,
                                                  float* __restrict__ ccur)
{
    int t = blockIdx.x * 256 + threadIdx.x;   // 933,888
    int col = t % 3072;
    int r = t / 3072;
    float val = 0.f;
    if (r < 300 && (col < 1024 || col >= 2048)) val = emb[(size_t)r * 1024 + (col & 1023)];
    A[t] = (f16)val;
    if (r < 300 && col < 1024) {
        hcur[(size_t)r * 1024 + col] = val;
        ccur[(size_t)r * 1024 + col] = 0.f;
    }
}

// attn = softmax(h @ ctx^T); readout = attn @ ctx -> Afce cols [1024,2048)
__global__ __launch_bounds__(256) void k_fce_attn(const float* __restrict__ hcur,
                                                  const float* __restrict__ ctx,
                                                  f16* __restrict__ A)
{
    __shared__ float hsd[1024];
    __shared__ float attn[32];
    int r = blockIdx.x;                       // 0..299
    int b = r / 75;
    int tid = threadIdx.x;
    for (int k = tid; k < 1024; k += 256) hsd[k] = hcur[(size_t)r * 1024 + k];
    __syncthreads();
    if (tid < 25) {
        const float* cr = ctx + (size_t)(b * 25 + tid) * 1024;
        float d = 0.f;
        for (int k = 0; k < 1024; ++k) d += hsd[k] * cr[k];
        attn[tid] = d;
    }
    __syncthreads();
    if (tid == 0) {
        float mx = -1e30f;
        for (int t = 0; t < 25; ++t) mx = fmaxf(mx, attn[t]);
        float sum = 0.f;
        for (int t = 0; t < 25; ++t) { float e = expf(attn[t] - mx); attn[t] = e; sum += e; }
        float inv = 1.f / sum;
        for (int t = 0; t < 25; ++t) attn[t] *= inv;
    }
    __syncthreads();
    for (int k = tid; k < 1024; k += 256) {
        float rd = 0.f;
        for (int t = 0; t < 25; ++t) rd += attn[t] * ctx[(size_t)(b * 25 + t) * 1024 + k];
        A[(size_t)r * 3072 + 1024 + k] = (f16)rd;
    }
}

// gates[304][4096] = A[304][3072] @ Bc^T + bC ; one wave = 16 rows x 64 cols.
// 96 K-steps of 16x16x32, software-pipelined by one stage: iteration k's four
// WMMAs run on registers while iteration k+1's A + 4xB tiles are in flight,
// so the compiler can issue partial loadcnt waits instead of draining to 0.
__global__ __launch_bounds__(256) void k_fce_gemm(const f16* __restrict__ A,
                                                  const f16* __restrict__ Bt,
                                                  const float* __restrict__ bC,
                                                  float* __restrict__ gates)
{
    int wid = (blockIdx.x * 256 + threadIdx.x) >> 5;   // 0..1215
    int lane = threadIdx.x & 31;
    int ntg = wid & 63;
    int mt = wid >> 6;
    int row = lane & 15;
    int hsel = lane >> 4;

    const f16* arow = A + (size_t)(mt * 16 + row) * 3072;
    const f16* brow = Bt + (size_t)(ntg * 64 + row) * 3072 + hsel * 16;

    v8f acc[4];
#pragma unroll
    for (int g = 0; g < 4; ++g)
#pragma unroll
        for (int i = 0; i < 8; ++i) acc[g][i] = 0.f;

    auto loadA = [&](int kk) -> v16h {
        const f16* pa = arow + (kk << 5) + hsel * 16;
        v8h lo = *(const v8h*)pa;
        v8h hi = *(const v8h*)(pa + 8);
        return __builtin_shufflevector(lo, hi,
            0, 1, 2, 3, 4, 5, 6, 7, 8, 9, 10, 11, 12, 13, 14, 15);
    };
    auto loadB = [&](int kk, int g) -> v16h {
        return *(const v16h*)(brow + (size_t)g * 16 * 3072 + (kk << 5));
    };

    v16h aop = loadA(0);
    v16h b0 = loadB(0, 0), b1 = loadB(0, 1), b2 = loadB(0, 2), b3 = loadB(0, 3);

    for (int kk = 0; kk < 95; ++kk) {
        __builtin_prefetch(arow + ((kk + 2) << 5), 0, 1);
        v16h aN = loadA(kk + 1);
        v16h n0 = loadB(kk + 1, 0);
        v16h n1 = loadB(kk + 1, 1);
        v16h n2 = loadB(kk + 1, 2);
        v16h n3 = loadB(kk + 1, 3);
        acc[0] = wmma_f16(aop, b0, acc[0]);
        acc[1] = wmma_f16(aop, b1, acc[1]);
        acc[2] = wmma_f16(aop, b2, acc[2]);
        acc[3] = wmma_f16(aop, b3, acc[3]);
        aop = aN; b0 = n0; b1 = n1; b2 = n2; b3 = n3;
    }
    acc[0] = wmma_f16(aop, b0, acc[0]);
    acc[1] = wmma_f16(aop, b1, acc[1]);
    acc[2] = wmma_f16(aop, b2, acc[2]);
    acc[3] = wmma_f16(aop, b3, acc[3]);

#pragma unroll
    for (int g = 0; g < 4; ++g) {
        int n = ntg * 64 + g * 16 + row;
        float bn = bC[n];
#pragma unroll
        for (int v = 0; v < 8; ++v) {
            int m = mt * 16 + v + hsel * 8;
            gates[(size_t)m * 4096 + n] = acc[g][v] + bn;
        }
    }
}

// LSTM cell elementwise + residual carry (h = cell_h + q_emb)
__global__ __launch_bounds__(256) void k_fce_cell(const float* __restrict__ gates,
                                                  const float* __restrict__ emb,
                                                  float* __restrict__ hcur,
                                                  float* __restrict__ ccur,
                                                  f16* __restrict__ A)
{
    int t = blockIdx.x * 256 + threadIdx.x;   // 307,200
    int r = t >> 10;
    int e = t & 1023;
    const float* g = gates + (size_t)r * 4096;
    float gi = g[e], gf = g[1024 + e], gg = g[2048 + e], go = g[3072 + e];
    float c = ccur[t];
    float cn = sigm(gf) * c + sigm(gi) * tanhf(gg);
    float hc = sigm(go) * tanhf(cn);
    float hcar = hc + emb[(size_t)r * 1024 + e];
    ccur[t] = cn;
    hcur[t] = hcar;
    A[(size_t)r * 3072 + 2048 + e] = (f16)hcar;
}

// Final: cosine sims (x100, relu), softmax over 25, class-sum over 5 shots, log
__global__ __launch_bounds__(256) void k_final(const float* __restrict__ hcur,
                                               const float* __restrict__ snorm,
                                               float* __restrict__ out)
{
    __shared__ float hsd[1024];
    __shared__ float red[256];
    __shared__ float sims[32];
    int r = blockIdx.x;                       // 0..299
    int b = r / 75;
    int tid = threadIdx.x;
    float ss = 0.f;
    for (int k = tid; k < 1024; k += 256) {
        float x = hcur[(size_t)r * 1024 + k];
        hsd[k] = x;
        ss += x * x;
    }
    red[tid] = ss;
    __syncthreads();
    for (int o = 128; o > 0; o >>= 1) {
        if (tid < o) red[tid] += red[tid + o];
        __syncthreads();
    }
    float inv = 1.f / fmaxf(sqrtf(red[0]), 1e-12f);
    if (tid < 25) {
        const float* sn = snorm + (size_t)(b * 25 + tid) * 1024;
        float d = 0.f;
        for (int k = 0; k < 1024; ++k) d += hsd[k] * sn[k];
        sims[tid] = fmaxf(d * inv, 0.f) * 100.f;
    }
    __syncthreads();
    if (tid == 0) {
        float mx = -1e30f;
        for (int t = 0; t < 25; ++t) mx = fmaxf(mx, sims[t]);
        float sum = 0.f, ex[25];
        for (int t = 0; t < 25; ++t) { ex[t] = expf(sims[t] - mx); sum += ex[t]; }
        float isum = 1.f / sum;
        for (int w = 0; w < 5; ++w) {
            float p = 0.f;
            for (int sh = 0; sh < 5; ++sh) p += ex[w * 5 + sh] * isum;
            out[r * 5 + w] = logf(p + 1e-6f);
        }
    }
}

// ---------------------------------------------------------------------------
// Host launcher
// ---------------------------------------------------------------------------

extern "C" void kernel_launch(void* const* d_in, const int* in_sizes, int n_in,
                              void* d_out, int out_size, void* d_ws, size_t ws_size,
                              hipStream_t stream)
{
    (void)in_sizes; (void)n_in; (void)out_size; (void)ws_size;

    const float* q_in = (const float*)d_in[0];
    const float* s_in = (const float*)d_in[1];
    const float *cw[4], *cb[4], *bg[4], *bb[4], *bm[4], *bv[4];
    for (int i = 0; i < 4; ++i) {
        int base = 2 + i * 6;
        cw[i] = (const float*)d_in[base];
        cb[i] = (const float*)d_in[base + 1];
        bg[i] = (const float*)d_in[base + 2];
        bb[i] = (const float*)d_in[base + 3];
        bm[i] = (const float*)d_in[base + 4];
        bv[i] = (const float*)d_in[base + 5];
    }
    const float* wf_ih = (const float*)d_in[26];
    const float* wf_hh = (const float*)d_in[27];
    const float* bf_ih = (const float*)d_in[28];
    const float* bf_hh = (const float*)d_in[29];
    const float* wr_ih = (const float*)d_in[30];
    const float* wr_hh = (const float*)d_in[31];
    const float* br_ih = (const float*)d_in[32];
    const float* br_hh = (const float*)d_in[33];
    const float* wc_ih = (const float*)d_in[34];
    const float* wc_hh = (const float*)d_in[35];
    const float* bc_ih = (const float*)d_in[36];
    const float* bc_hh = (const float*)d_in[37];
    float* out = (float*)d_out;
    char* ws = (char*)d_ws;

    // Workspace layout (all offsets multiples of 256B)
    constexpr size_t SZ_ACT1 = 400ull * 32 * 32 * 64 * 2;   // 52,428,800
    constexpr size_t SZ_ACT2 = 400ull * 16 * 16 * 64 * 2;   // 13,107,200
    constexpr size_t SZ_ACT3 = 400ull * 8 * 8 * 64 * 2;     //  3,276,800
    constexpr size_t SZ_ACT4 = 400ull * 4 * 4 * 64 * 2;     //    819,200
    constexpr size_t OFF_ACT1 = 0;
    constexpr size_t OFF_PRE2 = OFF_ACT1 + SZ_ACT1;
    constexpr size_t OFF_ACT2 = OFF_PRE2 + SZ_ACT1;
    constexpr size_t OFF_PRE3 = OFF_ACT1;                   // act1 dead after conv2
    constexpr size_t OFF_ACT3 = OFF_ACT2 + SZ_ACT2;
    constexpr size_t OFF_PRE4 = OFF_PRE2;                   // pre2 dead after pool2
    constexpr size_t OFF_ACT4 = OFF_ACT3 + SZ_ACT3;
    constexpr size_t OFF_EMB   = OFF_ACT4 + SZ_ACT4;        // f32 [400][1024]
    constexpr size_t OFF_WCONV = OFF_EMB + 400ull * 1024 * 4;       // f16 3x[9][64][64]
    constexpr size_t OFF_SCB   = OFF_WCONV + 3ull * 9 * 64 * 64 * 2; // f32 512
    constexpr size_t OFF_WTFI  = OFF_SCB + 2048;            // f16 [1024][4096] x4
    constexpr size_t SZ_WT = 1024ull * 4096 * 2;
    constexpr size_t OFF_WTFH = OFF_WTFI + SZ_WT;
    constexpr size_t OFF_WTRI = OFF_WTFH + SZ_WT;
    constexpr size_t OFF_WTRH = OFF_WTRI + SZ_WT;
    constexpr size_t OFF_BC    = OFF_WTRH + SZ_WT;          // f16 [4096][3072]
    constexpr size_t OFF_BF    = OFF_BC + 4096ull * 3072 * 2;
    constexpr size_t OFF_BR    = OFF_BF + 4096 * 4;
    constexpr size_t OFF_BCC   = OFF_BR + 4096 * 4;
    constexpr size_t OFF_HSCAN = OFF_BCC + 4096 * 4;        // f32 [2][8][1024]
    constexpr size_t OFF_CSCAN = OFF_HSCAN + 2ull * 8 * 1024 * 4;
    constexpr size_t OFF_HSEQ  = OFF_CSCAN + 8ull * 1024 * 4;   // f32 [8][25][1024]
    constexpr size_t OFF_CTX   = OFF_HSEQ + 8ull * 25 * 1024 * 4;
    constexpr size_t OFF_SNORM = OFF_CTX + 100ull * 1024 * 4;
    constexpr size_t OFF_AFCE  = OFF_SNORM + 100ull * 1024 * 4;  // f16 [304][3072]
    constexpr size_t OFF_GATES = OFF_AFCE + 304ull * 3072 * 2;   // f32 [304][4096]
    constexpr size_t OFF_HCUR  = OFF_GATES + 304ull * 4096 * 4;  // f32 [300][1024]
    constexpr size_t OFF_CCUR  = OFF_HCUR + 300ull * 1024 * 4;

    f16* act1 = (f16*)(ws + OFF_ACT1);
    f16* pre2 = (f16*)(ws + OFF_PRE2);
    f16* act2 = (f16*)(ws + OFF_ACT2);
    f16* pre3 = (f16*)(ws + OFF_PRE3);
    f16* act3 = (f16*)(ws + OFF_ACT3);
    f16* pre4 = (f16*)(ws + OFF_PRE4);
    f16* act4 = (f16*)(ws + OFF_ACT4);
    float* emb = (float*)(ws + OFF_EMB);
    f16* wconv = (f16*)(ws + OFF_WCONV);    // [blk][9][64][64], blk 0->conv2
    float* scb = (float*)(ws + OFF_SCB);
    f16* WtFih = (f16*)(ws + OFF_WTFI);
    f16* WtFhh = (f16*)(ws + OFF_WTFH);
    f16* WtRih = (f16*)(ws + OFF_WTRI);
    f16* WtRhh = (f16*)(ws + OFF_WTRH);
    f16* Bc = (f16*)(ws + OFF_BC);
    float* bF = (float*)(ws + OFF_BF);
    float* bR = (float*)(ws + OFF_BR);
    float* bC = (float*)(ws + OFF_BCC);
    float* hscan = (float*)(ws + OFF_HSCAN);
    float* cscan = (float*)(ws + OFF_CSCAN);
    float* hseq = (float*)(ws + OFF_HSEQ);
    float* ctx = (float*)(ws + OFF_CTX);
    float* snorm = (float*)(ws + OFF_SNORM);
    f16* Afce = (f16*)(ws + OFF_AFCE);
    float* gates = (float*)(ws + OFF_GATES);
    float* hcur = (float*)(ws + OFF_HCUR);
    float* ccur = (float*)(ws + OFF_CCUR);

    // --- prep ---
    k_prep_sb<<<1, 256, 0, stream>>>(
        cb[0], bg[0], bb[0], bm[0], bv[0], cb[1], bg[1], bb[1], bm[1], bv[1],
        cb[2], bg[2], bb[2], bm[2], bv[2], cb[3], bg[3], bb[3], bm[3], bv[3], scb);
    for (int i = 0; i < 3; ++i)
        k_prep_convw<<<144, 256, 0, stream>>>(cw[i + 1], wconv + (size_t)i * 9 * 64 * 64);
    k_transpose<<<16384, 256, 0, stream>>>(wf_ih, WtFih);
    k_transpose<<<16384, 256, 0, stream>>>(wf_hh, WtFhh);
    k_transpose<<<16384, 256, 0, stream>>>(wr_ih, WtRih);
    k_transpose<<<16384, 256, 0, stream>>>(wr_hh, WtRhh);
    k_prep_fcew<<<49152, 256, 0, stream>>>(wc_ih, wc_hh, Bc);
    k_prep_bias<<<48, 256, 0, stream>>>(bf_ih, bf_hh, br_ih, br_hh, bc_ih, bc_hh, bF, bR, bC);

    // --- encoder ---
    k_conv1<<<102400, 256, 0, stream>>>(q_in, s_in, cw[0], scb, act1);
    k_conv_wmma<32, 64><<<12800, 256, 0, stream>>>(act1, wconv, scb + 64, scb + 256 + 64, pre2);
    k_pool<32><<<25600, 256, 0, stream>>>(pre2, act2);
    k_conv_wmma<16, 16><<<3200, 256, 0, stream>>>(act2, wconv + 9 * 64 * 64,
                                                  scb + 128, scb + 256 + 128, pre3);
    k_pool<16><<<6400, 256, 0, stream>>>(pre3, act3);
    k_conv_wmma<8, 4><<<800, 256, 0, stream>>>(act3, wconv + 2 * 9 * 64 * 64,
                                               scb + 192, scb + 256 + 192, pre4);
    k_pool<8><<<1600, 256, 0, stream>>>(pre4, act4);
    k_flatten<<<1600, 256, 0, stream>>>(act4, emb);

    // --- support BiLSTM scan (double-buffered h) ---
    k_zero2<<<32, 256, 0, stream>>>(hscan, cscan);
    for (int s = 0; s < 25; ++s) {
        float* hr = hscan + (size_t)(s & 1) * 8192;
        float* hw = hscan + (size_t)((s + 1) & 1) * 8192;
        k_scan_step<<<8, 256, 0, stream>>>(emb, WtFih, WtFhh, WtRih, WtRhh,
                                           bF, bR, hr, hw, cscan, hseq, s);
    }
    k_ctxnorm<<<100, 256, 0, stream>>>(emb, hseq, ctx, snorm);

    // --- FCE: 25 steps of attention + WMMA LSTM-cell GEMM ---
    k_fce_init<<<3648, 256, 0, stream>>>(emb, Afce, hcur, ccur);
    for (int s = 0; s < 25; ++s) {
        k_fce_attn<<<300, 256, 0, stream>>>(hcur, ctx, Afce);
        k_fce_gemm<<<152, 256, 0, stream>>>(Afce, Bc, bC, gates);
        k_fce_cell<<<1200, 256, 0, stream>>>(gates, emb, hcur, ccur, Afce);
    }
    k_final<<<300, 256, 0, stream>>>(hcur, snorm, out);
}
